// FlaxGPTNeoSelfAttention_55327768707577
// MI455X (gfx1250) — compile-verified, matrix operand order checked by reading, not run on hardware
//
#include <hip/hip_runtime.h>
#include <hip/hip_bf16.h>

typedef __attribute__((ext_vector_type(16))) __bf16 v16bf;
typedef __attribute__((ext_vector_type(8)))  __bf16 v8bf;
typedef __attribute__((ext_vector_type(8)))  float  v8f;

#define Bsz   2
#define Ssz   2048
#define Dsz   2048
#define Hn    16
#define HDim  128
#define WIN   256
#define Msz   (Bsz * Ssz)   /* 4096 rows for all GEMMs */

__device__ __forceinline__ __bf16 f2bf(float f) { return (__bf16)f; }

__device__ __forceinline__ v8f vzero() {
  v8f z;
#pragma unroll
  for (int i = 0; i < 8; ++i) z[i] = 0.0f;
  return z;
}

// Load one 16x32 bf16 WMMA A/B fragment from row-major LDS tile.
// Lane l in [0,16): row = row0 + l, K chunks {0..7, 16..23}.
// Lane l in [16,32): row = row0 + (l-16), K chunks {8..15, 24..31}.
__device__ __forceinline__ v16bf load_frag(const __bf16* base, int row0, int pitch) {
  const int lane = threadIdx.x & 31;
  const int lr = lane & 15;
  const int hi = lane >> 4;
  const __bf16* p = base + (row0 + lr) * pitch + hi * 8;
  v8bf a = *(const v8bf*)(p);
  v8bf b = *(const v8bf*)(p + 16);
  v16bf f;
#pragma unroll
  for (int i = 0; i < 8; ++i) { f[i] = a[i]; f[i + 8] = b[i]; }
  return f;
}

// ---------------------------------------------------------------------------
// Kernel 1: fused QKV projection.  C_bf16 = bf16( X_f32 @ W_f32 )
// grid = (N/128, M/128, 3), block = 256 (8 waves).
// ---------------------------------------------------------------------------
__global__ __launch_bounds__(256)
void qkv_gemm_kernel(const float* __restrict__ X,
                     const float* __restrict__ Wq,
                     const float* __restrict__ Wk,
                     const float* __restrict__ Wv,
                     __bf16* __restrict__ Qo,
                     __bf16* __restrict__ Ko,
                     __bf16* __restrict__ Vo) {
  const int K = Dsz, N = Dsz;
  const float* W;
  __bf16* C;
  if (blockIdx.z == 0)      { W = Wq; C = Qo; }
  else if (blockIdx.z == 1) { W = Wk; C = Ko; }
  else                      { W = Wv; C = Vo; }

  const int n0 = blockIdx.x * 128;
  const int m0 = blockIdx.y * 128;

  __shared__ __bf16 As[128 * 40];   // A tile 128x32 bf16, pitch 40
  __shared__ __bf16 Bt[128 * 40];   // W tile transposed: [n][k], pitch 40

  const int t = threadIdx.x;
  const int w = t >> 5, lane = t & 31, lr = lane & 15, hi = lane >> 4;
  const int rm = (w >> 1) * 32;     // wave rows   (4 waves in M)
  const int cn = (w & 1) * 64;      // wave cols   (2 waves in N)

  v8f acc[2][4];
#pragma unroll
  for (int mi = 0; mi < 2; ++mi)
#pragma unroll
    for (int ni = 0; ni < 4; ++ni) acc[mi][ni] = vzero();

  for (int kt = 0; kt < K / 32; ++kt) {
    const int k0 = kt * 32;
    // Stage A: 128x32 fp32 -> bf16, 1024 float4 loads across 256 threads.
#pragma unroll
    for (int i = 0; i < 4; ++i) {
      const int v = t + i * 256;
      const int row = v >> 3, c4 = v & 7;
      const float4 f = *(const float4*)(X + (size_t)(m0 + row) * K + k0 + c4 * 4);
      __bf16* d = As + row * 40 + c4 * 4;
      d[0] = f2bf(f.x); d[1] = f2bf(f.y); d[2] = f2bf(f.z); d[3] = f2bf(f.w);
      if (kt + 1 < K / 32)   // prefetch next K tile (global_prefetch_b8)
        __builtin_prefetch(X + (size_t)(m0 + row) * K + k0 + 32 + c4 * 4, 0, 1);
    }
    // Stage W transposed: read coalesced row-major float4 along N, scatter to [n][k].
#pragma unroll
    for (int i = 0; i < 4; ++i) {
      const int v = t + i * 256;
      const int kk = v >> 5, n4 = v & 31;
      const float4 f = *(const float4*)(W + (size_t)(k0 + kk) * N + n0 + n4 * 4);
      Bt[(n4 * 4 + 0) * 40 + kk] = f2bf(f.x);
      Bt[(n4 * 4 + 1) * 40 + kk] = f2bf(f.y);
      Bt[(n4 * 4 + 2) * 40 + kk] = f2bf(f.z);
      Bt[(n4 * 4 + 3) * 40 + kk] = f2bf(f.w);
    }
    __syncthreads();

    v16bf af[2];
    af[0] = load_frag(As, rm, 40);
    af[1] = load_frag(As, rm + 16, 40);
#pragma unroll
    for (int ni = 0; ni < 4; ++ni) {
      const v16bf bfrag = load_frag(Bt, cn + ni * 16, 40);
#pragma unroll
      for (int mi = 0; mi < 2; ++mi)
        acc[mi][ni] = __builtin_amdgcn_wmma_f32_16x16x32_bf16(
            false, af[mi], false, bfrag, (short)0, acc[mi][ni], false, false);
    }
    __syncthreads();
  }

  // Store C layout: VGPR r -> rows r (lanes 0-15) / r+8 (lanes 16-31), N = lane&15.
#pragma unroll
  for (int mi = 0; mi < 2; ++mi)
#pragma unroll
    for (int ni = 0; ni < 4; ++ni)
#pragma unroll
      for (int r = 0; r < 8; ++r) {
        const int row = m0 + rm + mi * 16 + r + hi * 8;
        const int col = n0 + cn + ni * 16 + lr;
        C[(size_t)row * N + col] = f2bf(acc[mi][ni][r]);
      }
}

// ---------------------------------------------------------------------------
// Kernel 2: sliding-window flash attention (bf16 WMMA, fp32 softmax state).
// grid = (S/128, B*H), block = 256 (8 waves; wave w owns query rows w*16..+15).
// ---------------------------------------------------------------------------
__global__ __launch_bounds__(256)
void attn_kernel(const __bf16* __restrict__ Q,
                 const __bf16* __restrict__ Kmat,
                 const __bf16* __restrict__ V,
                 const int* __restrict__ amask,
                 __bf16* __restrict__ Ao) {
  const int PITCH = 136;
  __shared__ __bf16 Qs[128 * 136];   // Q tile [q][d]
  __shared__ __bf16 Ks[128 * 136];   // K tile [key][d]
  __shared__ __bf16 Vt[128 * 136];   // V tile transposed [d][key]
  __shared__ __bf16 Ps[128 * 136];   // probabilities [q][key], wave-private rows

  const int qb = blockIdx.x;               // query block (0..15)
  const int bh = blockIdx.y;               // 0..31
  const int b = bh >> 4, h = bh & 15;
  const int q0 = qb * 128;
  const int t = threadIdx.x, w = t >> 5, lane = t & 31, lr = lane & 15, hi = lane >> 4;
  const size_t rowbase = ((size_t)b * Ssz) * Dsz + (size_t)h * HDim;

  // Stage Q tile once (bf16 copy, 16B vectors).
#pragma unroll
  for (int i = 0; i < 8; ++i) {
    const int v = t + i * 256;
    const int row = v >> 4, c8 = v & 15;
    const uint4 u = *(const uint4*)(Q + rowbase + (size_t)(q0 + row) * Dsz + c8 * 8);
    *(uint4*)(Qs + row * PITCH + c8 * 8) = u;
  }

  float mrow[8], lrow[8];
  v8f oacc[8];
#pragma unroll
  for (int r = 0; r < 8; ++r) { mrow[r] = -1e30f; lrow[r] = 0.0f; }
#pragma unroll
  for (int n = 0; n < 8; ++n) oacc[n] = vzero();

  const int kbStart = (qb >= 2) ? (qb - 2) : 0;   // window 256 spans <=3 key blocks
  for (int kb = kbStart; kb <= qb; ++kb) {
    const int k0g = kb * 128;
    __syncthreads();   // previous iteration done with Ks/Vt (also covers Qs staging)
    // Stage K tile and transposed V tile.
#pragma unroll
    for (int i = 0; i < 8; ++i) {
      const int v = t + i * 256;
      const int row = v >> 4, c8 = v & 15;
      const uint4 uk = *(const uint4*)(Kmat + rowbase + (size_t)(k0g + row) * Dsz + c8 * 8);
      *(uint4*)(Ks + row * PITCH + c8 * 8) = uk;
      const uint4 uv = *(const uint4*)(V + rowbase + (size_t)(k0g + row) * Dsz + c8 * 8);
      const __bf16* pv = (const __bf16*)&uv;
#pragma unroll
      for (int j = 0; j < 8; ++j) Vt[(c8 * 8 + j) * PITCH + row] = pv[j];
    }
    __syncthreads();

    // S = Q @ K^T  (unscaled per GPT-Neo quirk), 16x16 tiles over 128 keys.
    v8f st[8];
#pragma unroll
    for (int n = 0; n < 8; ++n) st[n] = vzero();
#pragma unroll
    for (int kc = 0; kc < 4; ++kc) {
      const v16bf aq = load_frag(Qs + kc * 32, w * 16, PITCH);
#pragma unroll
      for (int n = 0; n < 8; ++n) {
        const v16bf bk = load_frag(Ks + kc * 32, n * 16, PITCH);
        st[n] = __builtin_amdgcn_wmma_f32_16x16x32_bf16(
            false, aq, false, bk, (short)0, st[n], false, false);
      }
    }

    // Band mask + attention_mask, then online softmax.
    float nm[8];
#pragma unroll
    for (int r = 0; r < 8; ++r) nm[r] = mrow[r];
#pragma unroll
    for (int n = 0; n < 8; ++n) {
      const int key = k0g + n * 16 + lr;
      const int av = amask[b * Ssz + key];
#pragma unroll
      for (int r = 0; r < 8; ++r) {
        const int q = q0 + w * 16 + r + hi * 8;
        const bool ok = (key <= q) && (q - key < WIN) && (av > 0);
        const float s = ok ? st[n][r] : -1e30f;
        st[n][r] = s;
        nm[r] = fmaxf(nm[r], s);
      }
    }
    // Row max across the 16-lane half (rows live per-half in wave32 C layout).
#pragma unroll
    for (int r = 0; r < 8; ++r) {
      float v0 = nm[r];
      v0 = fmaxf(v0, __shfl_xor(v0, 1, 32));
      v0 = fmaxf(v0, __shfl_xor(v0, 2, 32));
      v0 = fmaxf(v0, __shfl_xor(v0, 4, 32));
      v0 = fmaxf(v0, __shfl_xor(v0, 8, 32));
      nm[r] = v0;
    }
    float alpha[8], rs[8];
#pragma unroll
    for (int r = 0; r < 8; ++r) {
      alpha[r] = __expf(mrow[r] - nm[r]);
      mrow[r] = nm[r];
      rs[r] = 0.0f;
    }
    // P = exp(S - m), spill to LDS (wave-private rows) for re-load as A fragments.
#pragma unroll
    for (int n = 0; n < 8; ++n) {
#pragma unroll
      for (int r = 0; r < 8; ++r) {
        const float p = __expf(st[n][r] - mrow[r]);
        rs[r] += p;
        Ps[(w * 16 + r + hi * 8) * PITCH + n * 16 + lr] = f2bf(p);
      }
    }
#pragma unroll
    for (int r = 0; r < 8; ++r) {
      float v0 = rs[r];
      v0 += __shfl_xor(v0, 1, 32);
      v0 += __shfl_xor(v0, 2, 32);
      v0 += __shfl_xor(v0, 4, 32);
      v0 += __shfl_xor(v0, 8, 32);
      lrow[r] = lrow[r] * alpha[r] + v0;
    }
#pragma unroll
    for (int n = 0; n < 8; ++n)
#pragma unroll
      for (int r = 0; r < 8; ++r) oacc[n][r] *= alpha[r];

    // O += P @ V   (A = P rows, B = Vt rows = columns of V).
#pragma unroll
    for (int kc = 0; kc < 4; ++kc) {
      const v16bf ap = load_frag(Ps + kc * 32, w * 16, PITCH);
#pragma unroll
      for (int n = 0; n < 8; ++n) {
        const v16bf bv = load_frag(Vt + kc * 32, n * 16, PITCH);
        oacc[n] = __builtin_amdgcn_wmma_f32_16x16x32_bf16(
            false, ap, false, bv, (short)0, oacc[n], false, false);
      }
    }
  }

  // Normalize and store bf16 attention output in [b, s, h*HD + d] layout.
#pragma unroll
  for (int r = 0; r < 8; ++r) {
    const float inv = 1.0f / fmaxf(lrow[r], 1e-20f);
#pragma unroll
    for (int n = 0; n < 8; ++n) {
      const int q = q0 + w * 16 + r + hi * 8;
      const int dj = n * 16 + lr;
      Ao[rowbase + (size_t)q * Dsz + dj] = f2bf(oacc[n][r] * inv);
    }
  }
}

// ---------------------------------------------------------------------------
// Kernel 3: output projection.  out_f32 = A_bf16 @ Wo_f32 + bo
// ---------------------------------------------------------------------------
__global__ __launch_bounds__(256)
void oproj_gemm_kernel(const __bf16* __restrict__ A,
                       const float* __restrict__ W,
                       const float* __restrict__ bias,
                       float* __restrict__ C) {
  const int K = Dsz, N = Dsz;
  const int n0 = blockIdx.x * 128;
  const int m0 = blockIdx.y * 128;

  __shared__ __bf16 As[128 * 40];
  __shared__ __bf16 Bt[128 * 40];

  const int t = threadIdx.x;
  const int w = t >> 5, lane = t & 31, lr = lane & 15, hi = lane >> 4;
  const int rm = (w >> 1) * 32;
  const int cn = (w & 1) * 64;

  v8f acc[2][4];
#pragma unroll
  for (int mi = 0; mi < 2; ++mi)
#pragma unroll
    for (int ni = 0; ni < 4; ++ni) acc[mi][ni] = vzero();

  for (int kt = 0; kt < K / 32; ++kt) {
    const int k0 = kt * 32;
    // Stage A (already bf16): 1024 8-byte vectors.
#pragma unroll
    for (int i = 0; i < 4; ++i) {
      const int v = t + i * 256;
      const int row = v >> 3, c4 = v & 7;
      const uint2 u = *(const uint2*)(A + (size_t)(m0 + row) * K + k0 + c4 * 4);
      *(uint2*)(As + row * 40 + c4 * 4) = u;
    }
    // Stage Wo transposed to [n][k] bf16.
#pragma unroll
    for (int i = 0; i < 4; ++i) {
      const int v = t + i * 256;
      const int kk = v >> 5, n4 = v & 31;
      const float4 f = *(const float4*)(W + (size_t)(k0 + kk) * N + n0 + n4 * 4);
      Bt[(n4 * 4 + 0) * 40 + kk] = f2bf(f.x);
      Bt[(n4 * 4 + 1) * 40 + kk] = f2bf(f.y);
      Bt[(n4 * 4 + 2) * 40 + kk] = f2bf(f.z);
      Bt[(n4 * 4 + 3) * 40 + kk] = f2bf(f.w);
    }
    __syncthreads();

    v16bf af[2];
    af[0] = load_frag(As, rm, 40);
    af[1] = load_frag(As, rm + 16, 40);
#pragma unroll
    for (int ni = 0; ni < 4; ++ni) {
      const v16bf bfrag = load_frag(Bt, cn + ni * 16, 40);
#pragma unroll
      for (int mi = 0; mi < 2; ++mi)
        acc[mi][ni] = __builtin_amdgcn_wmma_f32_16x16x32_bf16(
            false, af[mi], false, bfrag, (short)0, acc[mi][ni], false, false);
    }
    __syncthreads();
  }

#pragma unroll
  for (int mi = 0; mi < 2; ++mi)
#pragma unroll
    for (int ni = 0; ni < 4; ++ni)
#pragma unroll
      for (int r = 0; r < 8; ++r) {
        const int row = m0 + rm + mi * 16 + r + hi * 8;
        const int col = n0 + cn + ni * 16 + lr;
        C[(size_t)row * N + col] = acc[mi][ni][r] + bias[col];
      }
}

// ---------------------------------------------------------------------------
extern "C" void kernel_launch(void* const* d_in, const int* in_sizes, int n_in,
                              void* d_out, int out_size, void* d_ws, size_t ws_size,
                              hipStream_t stream) {
  const float* X    = (const float*)d_in[0];   // hidden_states [B,S,D] f32
  const int*  amask = (const int*)d_in[1];     // attention_mask [B,S] i32
  const float* Wq   = (const float*)d_in[2];
  const float* Wk   = (const float*)d_in[3];
  const float* Wv   = (const float*)d_in[4];
  const float* Wo   = (const float*)d_in[5];
  const float* bo   = (const float*)d_in[6];
  float* out = (float*)d_out;                  // [B,S,D] f32

  const size_t elems = (size_t)Msz * Dsz;      // 8M elements per bf16 buffer
  __bf16* qbf = (__bf16*)d_ws;
  __bf16* kbf = qbf + elems;
  __bf16* vbf = kbf + elems;
  __bf16* abf = vbf + elems;

  dim3 blk(256, 1, 1);
  qkv_gemm_kernel<<<dim3(Dsz / 128, Msz / 128, 3), blk, 0, stream>>>(
      X, Wq, Wk, Wv, qbf, kbf, vbf);
  attn_kernel<<<dim3(Ssz / 128, Bsz * Hn, 1), blk, 0, stream>>>(
      qbf, kbf, vbf, amask, abf);
  oproj_gemm_kernel<<<dim3(Dsz / 128, Msz / 128, 1), blk, 0, stream>>>(
      abf, Wo, bo, out);
}
